// Model_Recursive_LSTM_v2_42571715838281
// MI455X (gfx1250) — compile-verified, matrix operand order checked by reading, not run on hardware
//
#include <hip/hip_runtime.h>
#include <math.h>

#define CNUM 16384

typedef __attribute__((ext_vector_type(16))) __bf16 v16bf;
typedef __attribute__((ext_vector_type(8)))  float  v8f;

__device__ __forceinline__ float elu1(float x){ return x > 0.f ? x : expm1f(x); }
__device__ __forceinline__ float sigm(float x){ return 1.f/(1.f+__expf(-x)); }

__device__ __forceinline__ unsigned short f2bf(float x){
  unsigned u = __float_as_uint(x);
  unsigned r = (u + 0x7FFFu + ((u >> 16) & 1u)) >> 16;   // round-to-nearest-even
  return (unsigned short)r;
}
__device__ __forceinline__ unsigned pack2bf(float a, float b){
  return (unsigned)f2bf(a) | ((unsigned)f2bf(b) << 16);
}

// ---------------------------------------------------------------------------
// WMMA GEMM:  Cout[M,N] = act( A[M,K] (row stride lda) @ W[N,K]^T + bias (+Cout if addC) )
// block = 256 threads = 8 waves; tile 128(M) x 64(N); each wave owns 2x2 16x16 tiles
// K step 32; bf16 operands (v_wmma_f32_16x16x32_bf16), f32 accumulation.
// Bounds handling is branch-free in staging: scalar fast path for interior
// tiles, clamped-address + select-zero for edge tiles (no exec divergence).
// ---------------------------------------------------------------------------
#define BM 128
#define BN 64
#define BK 32
#define LSTR 34   // LDS row stride (elements); 68B rows avoid bank conflicts

union FragU { v16bf v; unsigned u[8]; };
union AccU  { v8f  v; float    f[8]; };

__global__ __launch_bounds__(256)
void gemm_bias_act(const float* __restrict__ A, long lda,
                   const float* __restrict__ W,
                   const float* __restrict__ bias,
                   float* __restrict__ Cout, long ldc,
                   int M, int N, int K, int addC, int act)
{
  __shared__ unsigned short As[BM * LSTR];   // 8704 B
  __shared__ unsigned short Ws[BN * LSTR];   // 4352 B

  const int tid  = threadIdx.x;
  const int lane = tid & 31;
  const int wave = tid >> 5;
  const int wm   = wave >> 1;       // 0..3 -> 32-row slab
  const int wn   = wave & 1;        // 0..1 -> 32-col slab
  const int half = lane >> 4;       // 0/1
  const int r    = lane & 15;
  const int m0   = blockIdx.y * BM;
  const int n0   = blockIdx.x * BN;
  const bool fullM = (m0 + BM <= M);     // block-uniform
  const bool fullN = (n0 + BN <= N);     // block-uniform

  // C/D layout: element v -> row = v + 8*half, col = r (within a 16x16 tile)
  AccU acc[2][2];
  if (addC) {
    #pragma unroll
    for (int ti = 0; ti < 2; ++ti)
      #pragma unroll
      for (int tj = 0; tj < 2; ++tj)
        #pragma unroll
        for (int v = 0; v < 8; ++v) {
          int row = m0 + wm*32 + ti*16 + half*8 + v;
          int col = n0 + wn*32 + tj*16 + r;
          long rc = row < M ? row : (M-1);
          int  cc = col < N ? col : (N-1);
          float x = Cout[rc*ldc + cc];              // always-valid address
          acc[ti][tj].f[v] = (row < M && col < N) ? x : 0.f;
        }
  } else {
    #pragma unroll
    for (int ti = 0; ti < 2; ++ti)
      #pragma unroll
      for (int tj = 0; tj < 2; ++tj)
        #pragma unroll
        for (int v = 0; v < 8; ++v) acc[ti][tj].f[v] = 0.f;
  }

  const int kcA = (tid & 3) * 8;     // A: thread covers rows {t>>2, t>>2+64}, 8 k each
  const int rowA = tid >> 2;
  const int kcW = (tid & 3) * 8;     // W: thread covers row t>>2 (n), 8 k
  const int rowW = tid >> 2;         // 0..63

  for (int k0 = 0; k0 < K; k0 += BK) {
    const bool kfull = (k0 + BK <= K);   // uniform per iteration

    // ---- stage A tile: 128 rows x 32 k, f32 -> bf16, packed dword LDS stores
    #pragma unroll
    for (int rep = 0; rep < 2; ++rep) {
      int row = rowA + rep*64;
      long gr = (long)(m0 + row);
      if (fullM && kfull) {
        const float* base = A + gr*lda + k0 + kcA;
        #pragma unroll
        for (int p = 0; p < 4; ++p) {
          unsigned pk = pack2bf(base[2*p], base[2*p+1]);
          *(unsigned*)(&As[row*LSTR + kcA + 2*p]) = pk;
        }
      } else {
        long grc = gr < M ? gr : (M-1);
        const float* base = A + grc*lda + k0;
        #pragma unroll
        for (int p = 0; p < 4; ++p) {
          int k_a = kcA + 2*p, k_b = k_a + 1;
          int ka = (k0 + k_a < K) ? k_a : (K-1-k0);
          int kb = (k0 + k_b < K) ? k_b : (K-1-k0);
          float xa = base[ka], xb = base[kb];
          if (gr >= M || k0 + k_a >= K) xa = 0.f;
          if (gr >= M || k0 + k_b >= K) xb = 0.f;
          *(unsigned*)(&As[row*LSTR + kcA + 2*p]) = pack2bf(xa, xb);
        }
      }
    }
    // ---- stage W tile: 64 n x 32 k
    {
      long gn = (long)(n0 + rowW);
      if (fullN && kfull) {
        const float* base = W + gn*(long)K + k0 + kcW;
        #pragma unroll
        for (int p = 0; p < 4; ++p) {
          unsigned pk = pack2bf(base[2*p], base[2*p+1]);
          *(unsigned*)(&Ws[rowW*LSTR + kcW + 2*p]) = pk;
        }
      } else {
        long gnc = gn < N ? gn : (N-1);
        const float* base = W + gnc*(long)K + k0;
        #pragma unroll
        for (int p = 0; p < 4; ++p) {
          int k_a = kcW + 2*p, k_b = k_a + 1;
          int ka = (k0 + k_a < K) ? k_a : (K-1-k0);
          int kb = (k0 + k_b < K) ? k_b : (K-1-k0);
          float xa = base[ka], xb = base[kb];
          if (gn >= N || k0 + k_a >= K) xa = 0.f;
          if (gn >= N || k0 + k_b >= K) xb = 0.f;
          *(unsigned*)(&Ws[rowW*LSTR + kcW + 2*p]) = pack2bf(xa, xb);
        }
      }
    }
    __syncthreads();

    // ---- fragments per ISA 16-bit A/B layout:
    //  lanes 0-15 : K 0..7 (u0-3) and 16..23 (u4-7); lanes 16-31: K 8..15 / 24..31
    FragU af[2], bf[2];
    #pragma unroll
    for (int ti = 0; ti < 2; ++ti) {
      const int arow = (wm*32 + ti*16 + r) * LSTR;
      #pragma unroll
      for (int j = 0; j < 8; ++j) {
        int koff = (j < 4) ? (half*8 + 2*j) : (16 + half*8 + 2*(j-4));
        af[ti].u[j] = *(const unsigned*)(&As[arow + koff]);
      }
    }
    #pragma unroll
    for (int tj = 0; tj < 2; ++tj) {
      const int brow = (wn*32 + tj*16 + r) * LSTR;   // B[k,n] = W[n,k]: lane = column n
      #pragma unroll
      for (int j = 0; j < 8; ++j) {
        int koff = (j < 4) ? (half*8 + 2*j) : (16 + half*8 + 2*(j-4));
        bf[tj].u[j] = *(const unsigned*)(&Ws[brow + koff]);
      }
    }

    #pragma unroll
    for (int ti = 0; ti < 2; ++ti)
      #pragma unroll
      for (int tj = 0; tj < 2; ++tj)
        acc[ti][tj].v = __builtin_amdgcn_wmma_f32_16x16x32_bf16(
            false, af[ti].v, false, bf[tj].v,
            (short)0, acc[ti][tj].v, false, false);

    __syncthreads();
  }

  // ---- epilogue: bias + optional ELU
  #pragma unroll
  for (int ti = 0; ti < 2; ++ti)
    #pragma unroll
    for (int tj = 0; tj < 2; ++tj) {
      int col = n0 + wn*32 + tj*16 + r;
      if (col < N) {
        float b = bias ? bias[col] : 0.f;
        #pragma unroll
        for (int v = 0; v < 8; ++v) {
          int row = m0 + wm*32 + ti*16 + half*8 + v;
          if (row < M) {
            float x = acc[ti][tj].f[v] + b;
            if (act) x = elu1(x);
            Cout[(long)row*ldc + col] = x;
          }
        }
      }
    }
}

// ---------------------------------------------------------------------------
__global__ __launch_bounds__(256)
void zero_f32(float* p, long n)
{
  long i = (long)blockIdx.x * blockDim.x + threadIdx.x;
  if (i < n) p[i] = 0.f;
}

// gate order i,f,g,o (torch); g buffer [B,4H], h/c [B,H]
__global__ __launch_bounds__(256)
void lstm_pointwise(const float* __restrict__ g, float* __restrict__ h,
                    float* __restrict__ c, int B, int H)
{
  long i = (long)blockIdx.x * blockDim.x + threadIdx.x;
  if (i >= (long)B*H) return;
  int b = (int)(i / H), j = (int)(i % H);
  const float* gr = g + (long)b*4*H;
  float ig = sigm(gr[j]);
  float fg = sigm(gr[H + j]);
  float gg = tanhf(gr[2*H + j]);
  float og = sigm(gr[3*H + j]);
  float cn = fg*c[i] + ig*gg;
  c[i] = cn;
  h[i] = og * tanhf(cn);
}

// build comp-MLP input rows: [comps(776) | hf(100) | hb(100) | expr_emb(100)]
__global__ __launch_bounds__(256)
void concat_rows(const float* __restrict__ comps, const float* __restrict__ hf,
                 const float* __restrict__ hb, const float* __restrict__ ee,
                 float* __restrict__ out)
{
  long i = (long)blockIdx.x * blockDim.x + threadIdx.x;
  if (i >= (long)CNUM * 1076) return;
  int row = (int)(i / 1076), col = (int)(i % 1076);
  float v;
  if      (col <  776) v = comps[(long)row*776 + col];
  else if (col <  876) v = hf[(long)row*100 + col - 776];
  else if (col <  976) v = hb[(long)row*100 + col - 876];
  else                 v = ee[(long)row*100 + col - 976];
  out[i] = v;
}

// ---------------------------------------------------------------------------
// batch-1 tree tail: one block, 256 threads, shared-memory matvecs
// ---------------------------------------------------------------------------
struct TailParams {
  const float* ce;           // [C,180]
  const int*   comp_indices; // [4][8]
  const int*   loop_idx;     // [6]
  const float* loops;        // [6][33]
  const float *cl_Wih, *cl_Whh, *cl_bih, *cl_bhh;  // comps_lstm 720x180 ...
  const float *nl_Wih, *nl_Whh, *nl_bih, *nl_bhh;  // nodes_lstm
  const float *cm_W0, *cm_b0, *cm_W1, *cm_b1;      // concat_mlp 200x393, 180x200
  const float *rg_W0, *rg_b0, *rg_W1, *rg_b1;      // reg_mlp 200x180, 180x200
  const float *pred_W, *pred_b;
  const float *no_comps, *no_nodes, *root_iter;
  float* out;
};

__device__ void bk_matvec(const float* __restrict__ W, const float* __restrict__ b,
                          const float* __restrict__ x, float* __restrict__ y,
                          int outN, int inN, bool do_elu)
{
  for (int o = threadIdx.x; o < outN; o += blockDim.x) {
    const float* w = W + (long)o * inN;
    float s = b ? b[o] : 0.f;
    for (int k = 0; k < inN; ++k) s += w[k] * x[k];
    y[o] = do_elu ? elu1(s) : s;
  }
  __syncthreads();
}

__device__ void bk_lstm180(const float* __restrict__ seq, int T,
                           const float* __restrict__ Wih, const float* __restrict__ Whh,
                           const float* __restrict__ bih, const float* __restrict__ bhh,
                           float* h, float* c, float* g)
{
  for (int j = threadIdx.x; j < 180; j += blockDim.x) { h[j] = 0.f; c[j] = 0.f; }
  __syncthreads();
  for (int t = 0; t < T; ++t) {
    const float* xt = seq + t*180;
    for (int o = threadIdx.x; o < 720; o += blockDim.x) {
      const float* wx = Wih + (long)o*180;
      const float* wh = Whh + (long)o*180;
      float s = bih[o] + bhh[o];
      for (int k = 0; k < 180; ++k) s += wx[k]*xt[k] + wh[k]*h[k];
      g[o] = s;
    }
    __syncthreads();
    for (int j = threadIdx.x; j < 180; j += blockDim.x) {
      float ig = sigm(g[j]), fg = sigm(g[180+j]);
      float gg = tanhf(g[360+j]), og = sigm(g[540+j]);
      float cn = fg*c[j] + ig*gg;
      c[j] = cn;
      h[j] = og * tanhf(cn);
    }
    __syncthreads();
  }
}

__global__ __launch_bounds__(256)
void tail_kernel(TailParams P)
{
  __shared__ float seq[8*180];
  __shared__ float h[180], c[180], g[720];
  __shared__ float xcat[400];
  __shared__ float ytmp[200];
  __shared__ float leaves[4][180];
  __shared__ float rr[2][180];
  __shared__ float node[180];
  const int tid = threadIdx.x;

  auto node_state = [&](const float* nh, const float* ch, const float* lp, float* dst) {
    for (int i = tid; i < 393; i += blockDim.x) {
      float v;
      if (i < 180) v = nh[i]; else if (i < 360) v = ch[i-180]; else v = lp[i-360];
      xcat[i] = v;
    }
    __syncthreads();
    bk_matvec(P.cm_W0, P.cm_b0, xcat, ytmp, 200, 393, true);
    bk_matvec(P.cm_W1, P.cm_b1, ytmp, dst, 180, 200, true);
  };

  // 4 leaves
  for (int L = 0; L < 4; ++L) {
    for (int i = tid; i < 8*180; i += blockDim.x) {
      int s = i / 180, k = i % 180;
      int row = P.comp_indices[L*8 + s];
      seq[i] = P.ce[(long)row*180 + k];
    }
    __syncthreads();
    bk_lstm180(seq, 8, P.cl_Wih, P.cl_Whh, P.cl_bih, P.cl_bhh, h, c, g);
    node_state(P.no_nodes, h, P.loops + P.loop_idx[2+L]*33, leaves[L]);
  }
  // 2 internal nodes
  for (int I = 0; I < 2; ++I) {
    for (int i = tid; i < 360; i += blockDim.x)
      seq[i] = (i < 180) ? leaves[2*I][i] : leaves[2*I+1][i-180];
    __syncthreads();
    bk_lstm180(seq, 2, P.nl_Wih, P.nl_Whh, P.nl_bih, P.nl_bhh, h, c, g);
    node_state(h, P.no_comps, P.loops + P.loop_idx[I]*33, rr[I]);
  }
  // root
  for (int i = tid; i < 360; i += blockDim.x)
    seq[i] = (i < 180) ? rr[0][i] : rr[1][i-180];
  __syncthreads();
  bk_lstm180(seq, 2, P.nl_Wih, P.nl_Whh, P.nl_bih, P.nl_bhh, h, c, g);
  node_state(h, P.no_comps, P.root_iter, node);

  // reg MLP + prediction + ReLU
  bk_matvec(P.rg_W0, P.rg_b0, node, ytmp, 200, 180, true);
  bk_matvec(P.rg_W1, P.rg_b1, ytmp, xcat, 180, 200, true);
  if (tid == 0) {
    float s = P.pred_b[0];
    for (int k = 0; k < 180; ++k) s += P.pred_W[k] * xcat[k];
    P.out[0] = s > 0.f ? s : 0.f;
  }
}

// ---------------------------------------------------------------------------
extern "C" void kernel_launch(void* const* d_in, const int* in_sizes, int n_in,
                              void* d_out, int out_size, void* d_ws, size_t ws_size,
                              hipStream_t stream)
{
  const float* comps   = (const float*)d_in[0];
  const float* vectors = (const float*)d_in[1];
  const float* loops   = (const float*)d_in[2];
  const float* exprt   = (const float*)d_in[3];
  const int*   cidx    = (const int*)d_in[4];
  const int*   lidx    = (const int*)d_in[5];
  const float* eW0=(const float*)d_in[6];  const float* eb0=(const float*)d_in[7];
  const float* eW1=(const float*)d_in[8];  const float* eb1=(const float*)d_in[9];
  const float* eW2=(const float*)d_in[10]; const float* eb2=(const float*)d_in[11];
  const float* eW3=(const float*)d_in[12]; const float* eb3=(const float*)d_in[13];
  const float* encW=(const float*)d_in[14]; const float* encb=(const float*)d_in[15];
  const float* fWih=(const float*)d_in[16]; const float* fWhh=(const float*)d_in[17];
  const float* fbih=(const float*)d_in[18]; const float* fbhh=(const float*)d_in[19];
  const float* bWih=(const float*)d_in[20]; const float* bWhh=(const float*)d_in[21];
  const float* bbih=(const float*)d_in[22]; const float* bbhh=(const float*)d_in[23];
  const float* cW0=(const float*)d_in[24]; const float* cb0=(const float*)d_in[25];
  const float* cW1=(const float*)d_in[26]; const float* cb1=(const float*)d_in[27];
  const float* cW2=(const float*)d_in[28]; const float* cb2=(const float*)d_in[29];
  const float* cW3=(const float*)d_in[30]; const float* cb3=(const float*)d_in[31];
  const float* clWih=(const float*)d_in[32]; const float* clWhh=(const float*)d_in[33];
  const float* clbih=(const float*)d_in[34]; const float* clbhh=(const float*)d_in[35];
  const float* nlWih=(const float*)d_in[36]; const float* nlWhh=(const float*)d_in[37];
  const float* nlbih=(const float*)d_in[38]; const float* nlbhh=(const float*)d_in[39];
  const float* mW0=(const float*)d_in[40]; const float* mb0=(const float*)d_in[41];
  const float* mW1=(const float*)d_in[42]; const float* mb1=(const float*)d_in[43];
  const float* rW0=(const float*)d_in[44]; const float* rb0=(const float*)d_in[45];
  const float* rW1=(const float*)d_in[46]; const float* rb1=(const float*)d_in[47];
  const float* pW =(const float*)d_in[48]; const float* pb =(const float*)d_in[49];
  const float* noC=(const float*)d_in[50]; const float* noN=(const float*)d_in[51];
  const float* rootI=(const float*)d_in[52];

  // workspace partition (f32)
  char* wp = (char*)d_ws;
  auto walloc = [&](size_t nf)->float* {
    float* p = (float*)wp;
    wp += ((nf*sizeof(float) + 255) / 256) * 256;
    return p;
  };
  float* R0 = walloc((size_t)CNUM*1076);  // concat input / venc / ping
  float* R1 = walloc((size_t)CNUM*600);   // pong
  float* R2 = walloc((size_t)CNUM*100);   // expr_emb
  float* R3 = walloc((size_t)CNUM*100);   // hf (fwd running h)
  float* R4 = walloc((size_t)CNUM*100);   // hb (bwd running h)
  float* R5 = walloc((size_t)CNUM*100);   // c state
  float* R6 = walloc((size_t)CNUM*400);   // gates / pong
  float* R7 = walloc((size_t)CNUM*180);   // ce
  (void)ws_size; (void)in_sizes; (void)n_in; (void)out_size;

  dim3 blk(256);
  auto gemm = [&](const float* A, long lda, const float* W, const float* bias,
                  float* C, long ldc, int M, int N, int K, int addC, int act) {
    dim3 grid((N + BN - 1) / BN, (M + BM - 1) / BM);
    gemm_bias_act<<<grid, blk, 0, stream>>>(A, lda, W, bias, C, ldc, M, N, K, addC, act);
  };
  auto zero = [&](float* p, long n) {
    zero_f32<<<dim3((unsigned)((n + 255) / 256)), blk, 0, stream>>>(p, n);
  };

  // 1) expression embedding MLP (ne*el = 255 == padded width, use expr_tree directly)
  gemm(exprt, 255, eW0, eb0, R1, 600, CNUM, 600, 255, 0, 1);
  gemm(R1,   600, eW1, eb1, R0, 350, CNUM, 350, 600, 0, 1);
  gemm(R0,   350, eW2, eb2, R1, 250, CNUM, 250, 350, 0, 1);
  gemm(R1,   250, eW3, eb3, R2, 100, CNUM, 100, 250, 0, 1);

  // 2) vector encoder: venc[C*16,36] = vectors @ enc_W^T + enc_b   (no activation)
  gemm(vectors, 36, encW, encb, R0, 36, CNUM*16, 36, 36, 0, 0);

  // 3) forward LSTM over 16 steps (h -> R3)
  zero(R3, (long)CNUM*100); zero(R5, (long)CNUM*100);
  for (int t = 0; t < 16; ++t) {
    gemm(R0 + (long)t*36, 576, fWih, fbih, R6, 400, CNUM, 400,  36, 0, 0);
    gemm(R3, 100,              fWhh, fbhh, R6, 400, CNUM, 400, 100, 1, 0);
    lstm_pointwise<<<dim3((CNUM*100 + 255)/256), blk, 0, stream>>>(R6, R3, R5, CNUM, 100);
  }
  // 4) backward LSTM (h -> R4)
  zero(R4, (long)CNUM*100); zero(R5, (long)CNUM*100);
  for (int t = 0; t < 16; ++t) {
    int tt = 15 - t;
    gemm(R0 + (long)tt*36, 576, bWih, bbih, R6, 400, CNUM, 400,  36, 0, 0);
    gemm(R4, 100,               bWhh, bbhh, R6, 400, CNUM, 400, 100, 1, 0);
    lstm_pointwise<<<dim3((CNUM*100 + 255)/256), blk, 0, stream>>>(R6, R4, R5, CNUM, 100);
  }

  // 5) concat [comps | hf | hb | expr_emb] -> R0 [C,1076]   (venc/gates now dead)
  concat_rows<<<dim3((unsigned)(((long)CNUM*1076 + 255)/256)), blk, 0, stream>>>(
      comps, R3, R4, R2, R0);

  // 6) computation embedding MLP -> ce (R7)
  gemm(R0, 1076, cW0, cb0, R1, 600, CNUM, 600, 1076, 0, 1);
  gemm(R1,  600, cW1, cb1, R6, 350, CNUM, 350,  600, 0, 1);
  gemm(R6,  350, cW2, cb2, R1, 200, CNUM, 200,  350, 0, 1);
  gemm(R1,  200, cW3, cb3, R7, 180, CNUM, 180,  200, 0, 1);

  // 7) tree tail (batch 1) -> d_out[0]
  TailParams P;
  P.ce = R7; P.comp_indices = cidx; P.loop_idx = lidx; P.loops = loops;
  P.cl_Wih = clWih; P.cl_Whh = clWhh; P.cl_bih = clbih; P.cl_bhh = clbhh;
  P.nl_Wih = nlWih; P.nl_Whh = nlWhh; P.nl_bih = nlbih; P.nl_bhh = nlbhh;
  P.cm_W0 = mW0; P.cm_b0 = mb0; P.cm_W1 = mW1; P.cm_b1 = mb1;
  P.rg_W0 = rW0; P.rg_b0 = rb0; P.rg_W1 = rW1; P.rg_b1 = rb1;
  P.pred_W = pW; P.pred_b = pb;
  P.no_comps = noC; P.no_nodes = noN; P.root_iter = rootI;
  P.out = (float*)d_out;
  tail_kernel<<<dim3(1), blk, 0, stream>>>(P);
}
